// FeaturePlanes_73650099192285
// MI455X (gfx1250) — matrix operand order repaired and checked by reading, not compiled, not run on hardware
//
#include <hip/hip_runtime.h>

typedef _Float16 v8h  __attribute__((ext_vector_type(8)));
typedef _Float16 v16h __attribute__((ext_vector_type(16)));
typedef float    v8f  __attribute__((ext_vector_type(8)));

#define K_FEAT      24
#define K_THREADS   256     // 8 waves of 32
#define K_PTS       128     // points per block -> 8 tiles of 16, one per wave
#define K_FSTRIDE   200     // halves per point-feature row (>=192 + pad)
#define K_KBLK      96      // K-block staged per pass (3 WMMA k-steps)
#define K_WSTRIDE   104     // halves per staged-weight row (96 + 8 pad)
#define K_DIN       72
#define K_HID       168
#define K_HIDPAD    176
#define K_DOUT      53
#define K_DOUTPAD   64

static constexpr size_t kSmemBytes =
    (size_t)(2 * K_PTS * K_FSTRIDE) * 2 +        // bufA + bufB (f16)       102400 B
    (size_t)(K_HIDPAD * K_WSTRIDE) * 2 +         // weight K-block (f16)     36608 B
    (size_t)K_HIDPAD * 4;                        // bias (f32)                 704 B
                                                 // total 139712 B -> 2 blocks / WGP

// ---- stage one 96-wide K-block of a KxN weight matrix, transposed to [n][k] f16 ----
__device__ __forceinline__ void stage_w_block(const float* __restrict__ W,
                                              int K, int N, int Npad, int kOff,
                                              _Float16* __restrict__ wT,
                                              int tid, int nthreads) {
  for (int idx = tid; idx < Npad * K_KBLK; idx += nthreads) {
    int n = idx / K_KBLK;
    int kp = idx - n * K_KBLK;
    int k = kOff + kp;
    float v = (k < K && n < N) ? W[k * N + n] : 0.0f;
    wT[n * K_WSTRIDE + kp] = (_Float16)v;
  }
}

__device__ __forceinline__ void stage_bias(const float* __restrict__ b, int N, int Npad,
                                           float* __restrict__ biasS, int tid, int nthreads) {
  for (int n = tid; n < Npad; n += nthreads)
    biasS[n] = (n < N) ? b[n] : 0.0f;
}

// ---- bilinear sample of one plane (24 channels) into an LDS f16 row ----
__device__ __forceinline__ void sample_plane(const float* __restrict__ plane, int S,
                                             float a, float b,
                                             _Float16* __restrict__ dst) {
  a = fminf(fmaxf(a, 0.0f), 1.0f);
  b = fminf(fmaxf(b, 0.0f), 1.0f);
  float fa = a * (float)(S - 1);
  float fb = b * (float)(S - 1);
  int i0 = (int)floorf(fa); i0 = min(max(i0, 0), S - 2);
  int j0 = (int)floorf(fb); j0 = min(max(j0, 0), S - 2);
  float wa = fa - (float)i0;
  float wb = fb - (float)j0;
  float w00 = (1.0f - wa) * (1.0f - wb);
  float w10 = wa * (1.0f - wb);
  float w01 = (1.0f - wa) * wb;
  float w11 = wa * wb;
  const float* p00 = plane + (size_t)(i0 * S + j0) * K_FEAT;
  const float* p10 = p00 + (size_t)S * K_FEAT;
#pragma unroll
  for (int c = 0; c < K_FEAT; c += 4) {
    float4 a00 = *(const float4*)(p00 + c);
    float4 a01 = *(const float4*)(p00 + K_FEAT + c);
    float4 a10 = *(const float4*)(p10 + c);
    float4 a11 = *(const float4*)(p10 + K_FEAT + c);
    dst[c + 0] = (_Float16)(a00.x * w00 + a10.x * w10 + a01.x * w01 + a11.x * w11);
    dst[c + 1] = (_Float16)(a00.y * w00 + a10.y * w10 + a01.y * w01 + a11.y * w11);
    dst[c + 2] = (_Float16)(a00.z * w00 + a10.z * w10 + a01.z * w01 + a11.z * w11);
    dst[c + 3] = (_Float16)(a00.w * w00 + a10.w * w10 + a01.w * w01 + a11.w * w11);
  }
}

// ---- one 96-deep WMMA partial pass for a 16-point tile (3 k-steps x NTILES) ----
// A layout (16-bit 16x32): lanes 0-15 hold K[0..7]/K[16..23], lanes 16-31 K[8..15]/K[24..31]
// B layout (16-bit 32x16): lanes 0-15 hold K[0..7]/K[16..23], lanes 16-31 K[8..15]/K[24..31]
template <int NTILES>
__device__ __forceinline__ void mlp_partial(const _Float16* __restrict__ in,
                                            const _Float16* __restrict__ wT,
                                            int kInOff, int tileBase, int lane,
                                            v8f* __restrict__ acc) {
  const int m = lane & 15;
  const int koffA = (lane < 16) ? 0 : 8;
  const int koffB = (lane < 16) ? 0 : 8;
  v8h alo[3], ahi[3];
  const _Float16* arow = in + (tileBase + m) * K_FSTRIDE + kInOff;
#pragma unroll
  for (int ks = 0; ks < 3; ++ks) {
    alo[ks] = *(const v8h*)(arow + ks * 32 + koffA);
    ahi[ks] = *(const v8h*)(arow + ks * 32 + 16 + koffA);
  }
#pragma unroll
  for (int nt = 0; nt < NTILES; ++nt) {
    const _Float16* brow = wT + (nt * 16 + m) * K_WSTRIDE;
#pragma unroll
    for (int ks = 0; ks < 3; ++ks) {
      v8h blo = *(const v8h*)(brow + ks * 32 + koffB);
      v8h bhi = *(const v8h*)(brow + ks * 32 + 16 + koffB);
      v16h av = __builtin_shufflevector(alo[ks], ahi[ks],
                                        0, 1, 2, 3, 4, 5, 6, 7, 8, 9, 10, 11, 12, 13, 14, 15);
      v16h bv = __builtin_shufflevector(blo, bhi,
                                        0, 1, 2, 3, 4, 5, 6, 7, 8, 9, 10, 11, 12, 13, 14, 15);
      acc[nt] = __builtin_amdgcn_wmma_f32_16x16x32_f16(false, av, false, bv, (short)0,
                                                       acc[nt], false, false);
    }
  }
}

// ---- bias (+ optional ReLU), repack accumulators to f16 rows in LDS ----
template <int NTILES, bool RELU>
__device__ __forceinline__ void epilogue_store(const v8f* __restrict__ acc,
                                               const float* __restrict__ biasS,
                                               _Float16* __restrict__ out,
                                               int tileBase, int lane) {
  const int m = lane & 15;
#pragma unroll
  for (int nt = 0; nt < NTILES; ++nt) {
    float bias = biasS[nt * 16 + m];
#pragma unroll
    for (int r = 0; r < 8; ++r) {
      float v = acc[nt][r] + bias;
      if (RELU) v = fmaxf(v, 0.0f);
      int row = tileBase + ((lane < 16) ? r : r + 8);  // C/D layout: VGPR r -> M=r / r+8
      out[row * K_FSTRIDE + nt * 16 + m] = (_Float16)v;
    }
  }
}

__global__ __launch_bounds__(K_THREADS) void triplane_mlp_kernel(
    const float* __restrict__ x,
    const float* __restrict__ p0, const float* __restrict__ p1, const float* __restrict__ p2,
    const float* __restrict__ p3, const float* __restrict__ p4, const float* __restrict__ p5,
    const float* __restrict__ p6, const float* __restrict__ p7, const float* __restrict__ p8,
    const float* __restrict__ W1, const float* __restrict__ b1,
    const float* __restrict__ W2, const float* __restrict__ b2,
    const float* __restrict__ W3, const float* __restrict__ b3,
    float* __restrict__ out, int N) {
  extern __shared__ _Float16 smem[];
  _Float16* bufA = smem;
  _Float16* bufB = bufA + K_PTS * K_FSTRIDE;
  _Float16* wT = bufB + K_PTS * K_FSTRIDE;
  float* biasS = (float*)(wT + K_HIDPAD * K_WSTRIDE);

  const float* planes[9] = {p0, p1, p2, p3, p4, p5, p6, p7, p8};
  const int sizes[3] = {64, 128, 256};

  const int t = threadIdx.x;
  const int lane = t & 31;
  const int wave = t >> 5;
  const int tileBase = wave * 16;  // one 16-point tile per wave
  const int blockStart = blockIdx.x * K_PTS;
  const bool sampler = (t < K_PTS);

  // zero the K = [176,192) tail columns once (read by the k=96 passes, never written)
  for (int idx = t; idx < K_PTS * 16; idx += K_THREADS) {
    int row = idx >> 4;
    int k = 176 + (idx & 15);
    bufA[row * K_FSTRIDE + k] = (_Float16)0.0f;
    bufB[row * K_FSTRIDE + k] = (_Float16)0.0f;
  }

  float u = 0.0f, v = 0.0f, w = 0.0f;
  if (sampler) {
    int pc = min(blockStart + t, N - 1);
    u = x[pc * 3 + 0];
    v = x[pc * 3 + 1];
    w = x[pc * 3 + 2];
  }

  v8f outacc[4];
#pragma unroll
  for (int nt = 0; nt < 4; ++nt) outacc[nt] = {};

  for (int lvl = 0; lvl < 3; ++lvl) {
    const int S = sizes[lvl];

    // --- phase 1: threads 0..127 sample features; threads 128..255 stage W1 + b1 ---
    if (sampler) {
      _Float16* frow = bufA + t * K_FSTRIDE;
      sample_plane(planes[lvl * 3 + 0], S, u, v, frow);
      sample_plane(planes[lvl * 3 + 1], S, u, w, frow + K_FEAT);
      sample_plane(planes[lvl * 3 + 2], S, v, w, frow + 2 * K_FEAT);
#pragma unroll
      for (int k = K_DIN; k < K_KBLK; ++k) frow[k] = (_Float16)0.0f;
    } else {
      stage_w_block(W1 + lvl * K_DIN * K_HID, K_DIN, K_HID, K_HIDPAD, 0, wT,
                    t - K_PTS, K_THREADS - K_PTS);
      stage_bias(b1 + lvl * K_HID, K_HID, K_HIDPAD, biasS, t - K_PTS, K_THREADS - K_PTS);
    }
    __syncthreads();

    // --- layer 1: K = 72 (one 96-block) ---
    v8f h[11];
#pragma unroll
    for (int nt = 0; nt < 11; ++nt) h[nt] = {};
    mlp_partial<11>(bufA, wT, 0, tileBase, lane, h);
    epilogue_store<11, true>(h, biasS, bufB, tileBase, lane);
    __syncthreads();

    // --- layer 2: K = 168 (two 96-blocks), acc lives in registers across restage ---
    stage_w_block(W2 + lvl * K_HID * K_HID, K_HID, K_HID, K_HIDPAD, 0, wT, t, K_THREADS);
    stage_bias(b2 + lvl * K_HID, K_HID, K_HIDPAD, biasS, t, K_THREADS);
    __syncthreads();
#pragma unroll
    for (int nt = 0; nt < 11; ++nt) h[nt] = {};
    mlp_partial<11>(bufB, wT, 0, tileBase, lane, h);
    __syncthreads();
    stage_w_block(W2 + lvl * K_HID * K_HID, K_HID, K_HID, K_HIDPAD, K_KBLK, wT, t, K_THREADS);
    __syncthreads();
    mlp_partial<11>(bufB, wT, K_KBLK, tileBase, lane, h);
    epilogue_store<11, true>(h, biasS, bufA, tileBase, lane);
    __syncthreads();

    // --- layer 3: K = 168 (two 96-blocks), accumulate into register output ---
    stage_w_block(W3 + lvl * K_HID * K_DOUT, K_HID, K_DOUT, K_DOUTPAD, 0, wT, t, K_THREADS);
    stage_bias(b3 + lvl * K_DOUT, K_DOUT, K_DOUTPAD, biasS, t, K_THREADS);
    __syncthreads();
    v8f g[4];
#pragma unroll
    for (int nt = 0; nt < 4; ++nt) g[nt] = {};
    mlp_partial<4>(bufA, wT, 0, tileBase, lane, g);
    __syncthreads();
    stage_w_block(W3 + lvl * K_HID * K_DOUT, K_HID, K_DOUT, K_DOUTPAD, K_KBLK, wT, t, K_THREADS);
    __syncthreads();
    mlp_partial<4>(bufA, wT, K_KBLK, tileBase, lane, g);
    {
      const int m = lane & 15;
#pragma unroll
      for (int nt = 0; nt < 4; ++nt) {
        float bias = biasS[nt * 16 + m];
#pragma unroll
        for (int r = 0; r < 8; ++r) outacc[nt][r] += g[nt][r] + bias;
      }
    }
    __syncthreads();
  }

  // --- coalesced f32 stores, cols < 53 ---
#pragma unroll
  for (int nt = 0; nt < 4; ++nt) {
    int col = nt * 16 + (lane & 15);
    if (col < K_DOUT) {
#pragma unroll
      for (int r = 0; r < 8; ++r) {
        int row = blockStart + tileBase + ((lane < 16) ? r : r + 8);
        if (row < N) out[(size_t)row * K_DOUT + col] = outacc[nt][r];
      }
    }
  }
}

extern "C" void kernel_launch(void* const* d_in, const int* in_sizes, int n_in,
                              void* d_out, int out_size, void* d_ws, size_t ws_size,
                              hipStream_t stream) {
  (void)n_in; (void)out_size; (void)d_ws; (void)ws_size;
  const float* x  = (const float*)d_in[0];
  const float* p0 = (const float*)d_in[1];
  const float* p1 = (const float*)d_in[2];
  const float* p2 = (const float*)d_in[3];
  const float* p3 = (const float*)d_in[4];
  const float* p4 = (const float*)d_in[5];
  const float* p5 = (const float*)d_in[6];
  const float* p6 = (const float*)d_in[7];
  const float* p7 = (const float*)d_in[8];
  const float* p8 = (const float*)d_in[9];
  const float* W1 = (const float*)d_in[10];
  const float* b1 = (const float*)d_in[11];
  const float* W2 = (const float*)d_in[12];
  const float* b2 = (const float*)d_in[13];
  const float* W3 = (const float*)d_in[14];
  const float* b3 = (const float*)d_in[15];
  float* out = (float*)d_out;

  int N = in_sizes[0] / 3;
  int blocks = (N + K_PTS - 1) / K_PTS;

  hipFuncSetAttribute((const void*)triplane_mlp_kernel,
                      hipFuncAttributeMaxDynamicSharedMemorySize, (int)kSmemBytes);

  triplane_mlp_kernel<<<blocks, K_THREADS, kSmemBytes, stream>>>(
      x, p0, p1, p2, p3, p4, p5, p6, p7, p8, W1, b1, W2, b2, W3, b3, out, N);
}
